// LatticeLSTM_39745627357369
// MI455X (gfx1250) — matrix operand order repaired and checked by reading, not compiled, not run on hardware
//
#include <hip/hip_runtime.h>
#include <cstddef>
#include <cstdint>

// ---------------- problem sizes ----------------
#define BB  16      // batch
#define SS  512     // seq len
#define DD  128     // input dim
#define HH  256     // hidden
#define WWD 128     // word emb dim
#define G3  768     // 3*H
#define NWG 16      // serial-phase workgroups (H/16 column slices)

typedef __attribute__((ext_vector_type(16))) _Float16 v16h;
typedef __attribute__((ext_vector_type(8)))  _Float16 v8h;
typedef __attribute__((ext_vector_type(8)))  float    v8f;
typedef __attribute__((ext_vector_type(4)))  float    v4f;
typedef __attribute__((ext_vector_type(4)))  unsigned v4u;
typedef __attribute__((ext_vector_type(8)))  int      v8i;
typedef __attribute__((ext_vector_type(4)))  int      v4i;

#define DEV static __device__ __forceinline__

#ifndef __has_builtin
#define __has_builtin(x) 0
#endif
#if __has_builtin(__builtin_amdgcn_tensor_load_to_lds) && \
    __has_builtin(__builtin_amdgcn_s_wait_tensorcnt)
#define HAVE_TDM 1
#else
#define HAVE_TDM 0
#endif

// ---- device-pass feature probes (warnings land in stderr, build still OK) ----
#if defined(__AMDGCN__)
#if HAVE_TDM
#warning "probe: TDM builtins AVAILABLE (tensor_load_to_lds path compiled in)"
#else
#warning "probe: TDM builtins MISSING (manual LDS copy fallback in use)"
#endif
#if __has_builtin(__builtin_amdgcn_wmma_f32_16x16x4_f32)
#warning "probe: wmma_f32_16x16x4_f32 AVAILABLE"
#else
#warning "probe: wmma_f32_16x16x4_f32 missing"
#endif
#if __has_builtin(__builtin_amdgcn_global_load_async_to_lds_b128)
#warning "probe: global_load_async_to_lds_b128 builtin AVAILABLE"
#else
#warning "probe: global_load_async_to_lds_b128 builtin missing"
#endif
#if __has_builtin(__builtin_amdgcn_swmmac_f32_16x16x64_f16)
#warning "probe: swmmac_f32_16x16x64_f16 AVAILABLE"
#else
#warning "probe: swmmac_f32_16x16x64_f16 missing"
#endif
#if __has_builtin(__builtin_amdgcn_cluster_id_x)
#warning "probe: cluster_id_x AVAILABLE"
#else
#warning "probe: cluster_id_x missing"
#endif
#endif // __AMDGCN__

// ---- WMMA operand builders --------------------------------------------------
// A operand (16x32 f16): lane l -> m=l&15, hi=l>>4 ; a[j]=A[m][k0+hi*8+j],
// a[8+j]=A[m][k0+hi*8+16+j]   (ISA 16-bit A layout)
DEV v16h loadA_f16(const _Float16* base, int ld, int k0, int lane) {
  int m = lane & 15, hi = lane >> 4;
  const _Float16* p = base + (size_t)m * ld + k0 + hi * 8;
  v8h lo = *(const v8h*)(p);
  v8h hv = *(const v8h*)(p + 16);
  v16h a;
#pragma unroll
  for (int j = 0; j < 8; ++j) { a[j] = lo[j]; a[8 + j] = hv[j]; }
  return a;
}

// A operand built from a per-lane f32 row pointer (embedding gather)
DEV v16h loadA_f32row(const float* row, int k0, int lane) {
  int hi = lane >> 4;
  const float* p = row + k0 + hi * 8;
  v4f x0 = *(const v4f*)(p);
  v4f x1 = *(const v4f*)(p + 4);
  v4f x2 = *(const v4f*)(p + 16);
  v4f x3 = *(const v4f*)(p + 20);
  v16h a;
#pragma unroll
  for (int j = 0; j < 4; ++j) {
    a[j]      = (_Float16)x0[j];
    a[4 + j]  = (_Float16)x1[j];
    a[8 + j]  = (_Float16)x2[j];
    a[12 + j] = (_Float16)x3[j];
  }
  return a;
}

// B operand from pre-packed layout: 32 lanes * 16 contiguous halfs per tile
DEV v16h loadB_packed(const _Float16* tile, int lane) {
  const _Float16* p = tile + lane * 16;
  v8h lo = *(const v8h*)(p);
  v8h hv = *(const v8h*)(p + 8);
  v16h b;
#pragma unroll
  for (int j = 0; j < 8; ++j) { b[j] = lo[j]; b[8 + j] = hv[j]; }
  return b;
}

// C init from packed f16 tile layout [tile][lane][8]: one 16B load per lane
DEV v8f loadC_packed16(const _Float16* tile, int lane) {
  v8h cv = *(const v8h*)(tile + lane * 8);
  v8f c;
#pragma unroll
  for (int j = 0; j < 8; ++j) c[j] = (float)cv[j];
  return c;
}

DEV void storeStage(float* st /*16x16*/, v8f acc, int lane) {
  int n = lane & 15, hi = lane >> 4;
#pragma unroll
  for (int v = 0; v < 8; ++v) st[(v + 8 * hi) * 16 + n] = acc[v];
}

DEV float sigmf(float x) { return 1.0f / (1.0f + __expf(-x)); }
DEV float ftanh(float x) { return 1.0f - 2.0f / (__expf(2.0f * x) + 1.0f); }

// ---- Tensor Data Mover: 2-D strided tile -> contiguous LDS -----------------
// tile: tileRows rows of rowElts4 4-byte elements, row stride strideElts4.
#if HAVE_TDM
DEV void tdm_load_2d(unsigned ldsOff, const void* gaddr,
                     unsigned tileRows, unsigned rowElts4, unsigned strideElts4) {
  unsigned long long ga = (unsigned long long)(uintptr_t)gaddr;
  v4u g0;
  g0[0] = 1u;                                   // count=1, user descriptor
  g0[1] = ldsOff;                               // lds_addr (bytes)
  g0[2] = (unsigned)ga;                         // global_addr[31:0]
  g0[3] = (unsigned)((ga >> 32) & 0x01FFFFFFu)  // global_addr[56:32]
          | (2u << 30);                         // type=2 ("image")
  v8i g1;
  g1[0] = (int)(2u << 16);                                  // data_size=4B
  g1[1] = (int)((rowElts4 & 0xFFFFu) << 16);                // tensor_dim0[15:0]
  g1[2] = (int)(((rowElts4 >> 16) & 0xFFFFu)                // tensor_dim0[31:16]
                | ((tileRows & 0xFFFFu) << 16));            // tensor_dim1[15:0]
  g1[3] = (int)(((tileRows >> 16) & 0xFFFFu)                // tensor_dim1[31:16]
                | ((rowElts4 & 0xFFFFu) << 16));            // tile_dim0
  g1[4] = (int)(tileRows & 0xFFFFu);                        // tile_dim1 (dim2=0)
  g1[5] = (int)strideElts4;                                 // dim0_stride[31:0]
  g1[6] = 0;
  g1[7] = 0;                                                // dim1_stride unused
  v4i gz = {0, 0, 0, 0};
#if __clang_major__ >= 23
  v8i gz8 = {0, 0, 0, 0, 0, 0, 0, 0};
  __builtin_amdgcn_tensor_load_to_lds(g0, g1, gz, gz, gz8, 0);
#else
  __builtin_amdgcn_tensor_load_to_lds(g0, g1, gz, gz, 0);
#endif
}
#endif

// ---------------- zero fill --------------------------------------------------
__global__ void zero_f(float* p, size_t n) {
  size_t i = (size_t)blockIdx.x * blockDim.x + threadIdx.x;
  size_t stride = (size_t)gridDim.x * blockDim.x;
  for (; i < n; i += stride) p[i] = 0.0f;
}

// ---------------- x transpose+convert: xF16[(t*16+b)*128+k] -----------------
__global__ void cvt_x(const float* __restrict__ x, _Float16* __restrict__ xF16) {
  int i = blockIdx.x * blockDim.x + threadIdx.x;
  if (i >= SS * BB * DD) return;
  int k = i & (DD - 1);
  int rb = i >> 7;           // t*16+b
  int b = rb & 15, t = rb >> 4;
  xF16[i] = (_Float16)x[((size_t)b * SS + t) * DD + k];
}

// ---------------- pack f32 weight (K x N) into WMMA-B tiled f16 -------------
// packed[((kt*NT+nt)*32 + lane)*16 + j] = W[kt*32 + (lane>>4)*8 + (j<8 ? j : j+8)][nt*16 + (lane&15)]
__global__ void pack_b(const float* __restrict__ src, _Float16* __restrict__ dst,
                       int KT, int NT) {
  int wid = (blockIdx.x * blockDim.x + threadIdx.x) >> 5;
  int lane = threadIdx.x & 31;
  if (wid >= KT * NT) return;
  int kt = wid / NT, nt = wid % NT;
  int N = NT * 16;
  int n = lane & 15, krow = kt * 32 + (lane >> 4) * 8;
  const float* s = src + (size_t)krow * N + nt * 16 + n;
  _Float16* d = dst + ((size_t)wid * 32 + lane) * 16;
#pragma unroll
  for (int j = 0; j < 16; ++j) d[j] = (_Float16)s[(size_t)(j < 8 ? j : j + 8) * N];
}

// ---------------- parallel prep GEMMs (carry-independent terms) -------------
// Output in packed WMMA-C tile layout: out[((mt*NT+nt)*32 + lane)*8 + v] (f16)
// mode 0: G1 = x @ Wih + b   mode 1: A1 = x @ aWih + ab
// mode 2: G2 = emb[wid] @ wWih + wb (f32 gather A)
__global__ __launch_bounds__(128)
void prep_gemm(const _Float16* __restrict__ xF16,
               const float* __restrict__ emb, const int* __restrict__ word_ids,
               const _Float16* __restrict__ pB, const float* __restrict__ bias,
               _Float16* __restrict__ outp, int NT, int mode) {
  int wid = (blockIdx.x * blockDim.x + threadIdx.x) >> 5;
  int lane = threadIdx.x & 31;
  if (wid >= (SS * BB / 16) * NT) return;
  int mt = wid / NT, nt = wid % NT;

  v8f acc = {};
  if (mode == 2) {            // uniform branch, hoisted out of the K-loop
    int m = lane & 15;
    int row = mt * 16 + m;
    int b = row & 15, t = row >> 4;
    const float* eRow = emb + (size_t)word_ids[b * SS + t] * WWD;
#pragma unroll
    for (int kt = 0; kt < 4; ++kt) {   // K = 128
      v16h a  = loadA_f32row(eRow, kt * 32, lane);
      v16h bm = loadB_packed(pB + ((size_t)(kt * NT + nt) * 32) * 16, lane);
      acc = __builtin_amdgcn_wmma_f32_16x16x32_f16(false, a, false, bm,
                                                   (short)0, acc, false, false);
    }
  } else {
    const _Float16* Abase = xF16 + (size_t)mt * 16 * DD;
#pragma unroll
    for (int kt = 0; kt < 4; ++kt) {   // K = 128
      v16h a  = loadA_f16(Abase, DD, kt * 32, lane);
      v16h bm = loadB_packed(pB + ((size_t)(kt * NT + nt) * 32) * 16, lane);
      acc = __builtin_amdgcn_wmma_f32_16x16x32_f16(false, a, false, bm,
                                                   (short)0, acc, false, false);
    }
  }
  float bv = bias[nt * 16 + (lane & 15)];
  _Float16* d = outp + ((size_t)wid * 32 + lane) * 8;
#pragma unroll
  for (int v = 0; v < 8; ++v) d[v] = (_Float16)(acc[v] + bv);
}

// ---------------- inter-workgroup generation barrier ------------------------
DEV void gbar(unsigned* bar) {
  __syncthreads();
  if (threadIdx.x == 0) {
    __threadfence();
    unsigned* cnt = bar;
    unsigned* gen = bar + 64;   // separate cache line
    unsigned g = __hip_atomic_load(gen, __ATOMIC_RELAXED, __HIP_MEMORY_SCOPE_AGENT);
    unsigned a = __hip_atomic_fetch_add(cnt, 1u, __ATOMIC_ACQ_REL, __HIP_MEMORY_SCOPE_AGENT);
    if (a == NWG - 1) {
      __hip_atomic_store(cnt, 0u, __ATOMIC_RELAXED, __HIP_MEMORY_SCOPE_AGENT);
      __hip_atomic_fetch_add(gen, 1u, __ATOMIC_RELEASE, __HIP_MEMORY_SCOPE_AGENT);
    } else {
      while (__hip_atomic_load(gen, __ATOMIC_ACQUIRE, __HIP_MEMORY_SCOPE_AGENT) == g)
        __builtin_amdgcn_s_sleep(2);
    }
    __threadfence();
  }
  __syncthreads();
}

// ---------------- serial recurrent scan: 16 WGs x 128 thr (4 waves) ---------
// WG r owns hidden columns [16r,16r+16). LDS holds its Whh/wWhh/aWhh slices,
// staged by the Tensor Data Mover. LDS B-tile row order follows the TDM's
// 2-D walk: global row (kt*48 + w*16 + r) lands at LDS row y = 3*kt + w.
__global__ __launch_bounds__(128)
void lattice_serial(const _Float16* __restrict__ pWhh,
                    const _Float16* __restrict__ paWhh,
                    const _Float16* __restrict__ pwWhh,
                    const _Float16* __restrict__ G1,   // packed-C tiles (S*48)
                    const _Float16* __restrict__ A1,   // packed-C tiles (S*16)
                    const _Float16* __restrict__ G2,   // packed-C tiles (S*48)
                    const int* __restrict__ lengths,
                    _Float16* __restrict__ hbuf,       // 2 x (16x256) ping-pong
                    float* __restrict__ bufF32,        // (S,16,256)
                    _Float16* __restrict__ bufF16,     // (S,16,256)
                    float* __restrict__ h_out,
                    float* __restrict__ c_out,
                    unsigned* __restrict__ bar) {
  __shared__ __attribute__((aligned(16))) _Float16 sWhh [24 * 512]; // 24KB
  __shared__ __attribute__((aligned(16))) _Float16 sWWhh[24 * 512]; // 24KB
  __shared__ __attribute__((aligned(16))) _Float16 saWhh[ 8 * 512]; // 8KB
  __shared__ float stage[4][16][16];                                // 4KB
  __shared__ float c1s[16][16];                                     // 1KB

  const int r = blockIdx.x;          // column-slice owner
  const int tid = threadIdx.x;
  const int w = tid >> 5, lane = tid & 31;
  const int colsBase = r * 16;

  // --- prologue: stage this WG's weight slices into LDS ---
#if HAVE_TDM
  if (w == 0) {
    // rows r + 16*y of the packed arrays; 1KB rows, 16KB stride, 4B elements
    tdm_load_2d((unsigned)(uintptr_t)(void*)sWhh,
                pWhh + (size_t)r * 512, 24u, 256u, 4096u);
    tdm_load_2d((unsigned)(uintptr_t)(void*)sWWhh,
                pwWhh + (size_t)r * 512, 24u, 256u, 4096u);
    tdm_load_2d((unsigned)(uintptr_t)(void*)saWhh,
                paWhh + (size_t)r * 512, 8u, 256u, 4096u);
    __builtin_amdgcn_s_wait_tensorcnt(0);
  }
#else
  for (int i = tid; i < 24 * 512; i += 128) {
    int e = i & 511, y = i >> 9;
    int kt = y / 3, w3 = y % 3;     // LDS row y = 3*kt + w3
    size_t g = ((size_t)(kt * 48 + w3 * 16 + r) * 512) + e;
    sWhh[i]  = pWhh[g];
    sWWhh[i] = pwWhh[g];
  }
  for (int i = tid; i < 8 * 512; i += 128) {
    int e = i & 511, kt = i >> 9;
    saWhh[i] = paWhh[((size_t)(kt * 16 + r) * 512) + e];
  }
#endif
  __syncthreads();
  gbar(bar);

  for (int t = 0; t < SS; ++t) {
    const _Float16* hR  = hbuf + (size_t)(t & 1) * (BB * HH);        // h_{t-1}
    _Float16*       hWp = hbuf + (size_t)((t + 1) & 1) * (BB * HH);  // h_t

    // ---- phase A: i/o/g (waves 0..2) and alpha (wave 3) -------------------
    {
      const _Float16* Abase;
      const _Float16* Bbase;
      const _Float16* Ctile;
      size_t bstep;
      if (w < 3) {
        Abase = hR;
        Bbase = sWhh + (size_t)w * 512;
        bstep = 3 * 512;
        Ctile = G1 + ((size_t)(t * 48 + w * 16 + r) * 32) * 8;
      } else {
        Abase = bufF16 + (size_t)t * BB * HH;
        Bbase = saWhh;
        bstep = 512;
        Ctile = A1 + ((size_t)(t * 16 + r) * 32) * 8;
      }
      v8f acc = loadC_packed16(Ctile, lane);
#pragma unroll
      for (int kt = 0; kt < 8; ++kt) {   // K = 256
        v16h a  = loadA_f16(Abase, HH, kt * 32, lane);
        v16h bm = loadB_packed(Bbase + (size_t)kt * bstep, lane);
        acc = __builtin_amdgcn_wmma_f32_16x16x32_f16(false, a, false, bm,
                                                     (short)0, acc, false, false);
      }
      storeStage(&stage[w][0][0], acc, lane);
    }

    // ---- prefetch next step's operands while results settle ---------------
    if (t + 1 < SS) {
      if (w < 3) {
        __builtin_prefetch(G1 + ((size_t)((t + 1) * 48 + w * 16 + r) * 32 + lane) * 8, 0, 1);
        __builtin_prefetch(G2 + ((size_t)((t + 1) * 48 + w * 16 + r) * 32 + lane) * 8, 0, 1);
      } else {
        __builtin_prefetch(A1 + ((size_t)((t + 1) * 16 + r) * 32 + lane) * 8, 0, 1);
        __builtin_prefetch(bufF16 + (size_t)(t + 1) * BB * HH + lane * 128, 0, 1);
      }
    }
    __syncthreads();

    // ---- elementwise A: multi-input cell (256 elems / 128 threads) --------
#pragma unroll
    for (int e2 = 0; e2 < 2; ++e2) {
      int e = tid + e2 * 128;
      int m = e & 15, n = e >> 4;
      float iR = stage[0][m][n], oR = stage[1][m][n];
      float gR = stage[2][m][n], aR = stage[3][m][n];
      float cin = bufF32[((size_t)t * BB + m) * HH + colsBase + n];
      float ei = __expf(sigmf(iR)), ea = __expf(sigmf(aR));
      float gg = ftanh(gR);
      float c1 = (ei * gg + ea * cin) / (ei + ea);
      float h1 = sigmf(oR) * ftanh(c1);
      c1s[m][n] = c1;
      size_t oidx = ((size_t)t * BB + m) * HH + colsBase + n;
      h_out[oidx] = h1;
      c_out[oidx] = c1;
      hWp[m * HH + colsBase + n] = (_Float16)h1;
    }
    gbar(bar);   // publish full h_t to all slices

    // ---- phase B: word-LSTM gates f/i2/g2 (waves 0..2) --------------------
    if (w < 3) {
      const _Float16* Bbase = sWWhh + (size_t)w * 512;
      v8f acc = loadC_packed16(G2 + ((size_t)(t * 48 + w * 16 + r) * 32) * 8, lane);
#pragma unroll
      for (int kt = 0; kt < 8; ++kt) {
        v16h a  = loadA_f16(hWp, HH, kt * 32, lane);
        v16h bm = loadB_packed(Bbase + (size_t)kt * (3 * 512), lane);
        acc = __builtin_amdgcn_wmma_f32_16x16x32_f16(false, a, false, bm,
                                                     (short)0, acc, false, false);
      }
      storeStage(&stage[w][0][0], acc, lane);
    }
    __syncthreads();

    // ---- elementwise B: ct + scatter to buf[t+len-1] ----------------------
#pragma unroll
    for (int e2 = 0; e2 < 2; ++e2) {
      int e = tid + e2 * 128;
      int m = e & 15, n = e >> 4;
      float fR = stage[0][m][n], i2R = stage[1][m][n], g2R = stage[2][m][n];
      float c1 = c1s[m][n];
      float ct = sigmf(fR) * c1 + sigmf(i2R) * ftanh(g2R);
      int lng = lengths[m * SS + t];
      int dst = t + lng - 1;
      size_t bidx = ((size_t)dst * BB + m) * HH + colsBase + n;
      bufF32[bidx] = ct;
      bufF16[bidx] = (_Float16)ct;
    }
    gbar(bar);   // step boundary: buf writes visible to step t+1
  }
}

// ---------------- launcher ---------------------------------------------------
extern "C" void kernel_launch(void* const* d_in, const int* in_sizes, int n_in,
                              void* d_out, int out_size, void* d_ws, size_t ws_size,
                              hipStream_t stream) {
  const float* x    = (const float*)d_in[0];
  const int*   wid  = (const int*)d_in[1];
  const int*   lens = (const int*)d_in[2];
  const float* Wih  = (const float*)d_in[3];
  const float* Whh  = (const float*)d_in[4];
  const float* bb   = (const float*)d_in[5];
  const float* aWih = (const float*)d_in[6];
  const float* aWhh = (const float*)d_in[7];
  const float* ab   = (const float*)d_in[8];
  const float* wWih = (const float*)d_in[9];
  const float* wWhh = (const float*)d_in[10];
  const float* wb   = (const float*)d_in[11];
  const float* emb  = (const float*)d_in[12];

  // ---- workspace carve-up (256B aligned) ----
  char* base = (char*)d_ws;
  size_t off = 0;
  auto carve = [&](size_t bytes) {
    char* p = base + off;
    off = (off + bytes + 255) & ~(size_t)255;
    return p;
  };
  unsigned*  bar    = (unsigned*)carve(512);
  _Float16*  hbuf   = (_Float16*)carve(2 * BB * HH * sizeof(_Float16));
  _Float16*  xF16   = (_Float16*)carve((size_t)SS * BB * DD * sizeof(_Float16));
  float*     bufF32 = (float*)   carve((size_t)SS * BB * HH * sizeof(float));
  _Float16*  bufF16 = (_Float16*)carve((size_t)SS * BB * HH * sizeof(_Float16));
  _Float16*  G1pre  = (_Float16*)carve((size_t)SS * BB * G3 * sizeof(_Float16));
  _Float16*  G2pre  = (_Float16*)carve((size_t)SS * BB * G3 * sizeof(_Float16));
  _Float16*  A1pre  = (_Float16*)carve((size_t)SS * BB * HH * sizeof(_Float16));
  _Float16*  pWih   = (_Float16*)carve((size_t)DD * G3 * sizeof(_Float16));
  _Float16*  pWhh   = (_Float16*)carve((size_t)HH * G3 * sizeof(_Float16));
  _Float16*  paWih  = (_Float16*)carve((size_t)DD * HH * sizeof(_Float16));
  _Float16*  paWhh  = (_Float16*)carve((size_t)HH * HH * sizeof(_Float16));
  _Float16*  pwWih  = (_Float16*)carve((size_t)WWD * G3 * sizeof(_Float16));
  _Float16*  pwWhh  = (_Float16*)carve((size_t)HH * G3 * sizeof(_Float16));

  float* h_out = (float*)d_out;
  float* c_out = (float*)d_out + (size_t)SS * BB * HH;

  // ---- init: zero buf / h / barrier ----
  zero_f<<<2048, 256, 0, stream>>>(bufF32, (size_t)SS * BB * HH);
  zero_f<<<1024, 256, 0, stream>>>((float*)bufF16, (size_t)SS * BB * HH / 2);
  zero_f<<<4, 256, 0, stream>>>((float*)hbuf, (size_t)2 * BB * HH / 2);
  zero_f<<<1, 128, 0, stream>>>((float*)bar, 128);

  // ---- convert x to f16 in (t,b,k) order ----
  cvt_x<<<(SS * BB * DD + 255) / 256, 256, 0, stream>>>(x, xF16);

  // ---- pack all weights into WMMA-B tiled f16 layout ----
  pack_b<<<48, 128, 0, stream>>>(Wih,  pWih,  4, 48);
  pack_b<<<96, 128, 0, stream>>>(Whh,  pWhh,  8, 48);
  pack_b<<<16, 128, 0, stream>>>(aWih, paWih, 4, 16);
  pack_b<<<32, 128, 0, stream>>>(aWhh, paWhh, 8, 16);
  pack_b<<<48, 128, 0, stream>>>(wWih, pwWih, 4, 48);
  pack_b<<<96, 128, 0, stream>>>(wWhh, pwWhh, 8, 48);

  // ---- fully parallel carry-independent GEMMs ----
  prep_gemm<<<6144, 128, 0, stream>>>(xF16, nullptr, nullptr, pWih,  bb, G1pre, 48, 0);
  prep_gemm<<<2048, 128, 0, stream>>>(xF16, nullptr, nullptr, paWih, ab, A1pre, 16, 1);
  prep_gemm<<<6144, 128, 0, stream>>>(nullptr, emb, wid,      pwWih, wb, G2pre, 48, 2);

  // ---- serial recurrent scan: 16 cooperating WGs ----
  lattice_serial<<<NWG, 128, 0, stream>>>(pWhh, paWhh, pwWhh, G1pre, A1pre, G2pre,
                                          lens, hbuf, bufF32, bufF16,
                                          h_out, c_out, bar);
}
